// BasicBlock_1w1a_88639535055445
// MI455X (gfx1250) — 8-bit matrix operand order repaired and checked by reading, compiled, not run on hardware
//
#include <hip/hip_runtime.h>


typedef __attribute__((ext_vector_type(8))) int v8i;
typedef int vi4 __attribute__((__vector_size__(16)));

#if __has_builtin(__builtin_amdgcn_global_load_async_to_lds_b128)
#define HAVE_ASYNC_LDS 1
typedef __attribute__((address_space(1))) vi4* g4p;
typedef __attribute__((address_space(3))) vi4* l4p;
#else
#define HAVE_ASYNC_LDS 0
#endif

#define N_   32
#define C_   256
#define H_   56
#define W_   56
#define HW_  (H_ * W_)
#define NHW_ (N_ * HW_)
#define CNTF ((float)NHW_)
#define EPS_ 1e-5f
#define SC_  0.5f

#define SZ_XBIN  ((size_t)NHW_ * C_)
#define SZ_WB    ((size_t)9 * C_ * C_)
#define OFF_XBIN  ((size_t)0)
#define OFF_A2BIN (OFF_XBIN + SZ_XBIN)
#define OFF_WB1   (OFF_A2BIN + SZ_XBIN)
#define OFF_WB2   (OFF_WB1 + SZ_WB)
#define OFF_STATS (OFF_WB2 + SZ_WB)
#define OFF_ZERO  (OFF_STATS + 4096)
#define OFF_Y32   (OFF_ZERO + 256)

#define LDSP 80

__global__ void k_init(float* stats, int* zp) {
  int i = blockIdx.x * blockDim.x + threadIdx.x;
  if (i < 4 * C_) stats[i] = 0.0f;
  if (i >= 4 * C_ && i < 4 * C_ + 64) zp[i - 4 * C_] = 0;
}

__global__ void k_binarize_x(const float* __restrict__ x, char* __restrict__ xb) {
  int i = blockIdx.x * blockDim.x + threadIdx.x;
  if (i >= NHW_ * C_) return;
  int c = i & (C_ - 1);
  int p = i >> 8;
  int w = p % W_;
  int h = (p / W_) % H_;
  int n = p / HW_;
  float v = x[((n * C_ + c) * H_ + h) * W_ + w];
  xb[i] = (v > 0.f) ? 1 : ((v < 0.f) ? -1 : 0);
}

__global__ void k_binarize_w(const float* __restrict__ w, char* __restrict__ wb) {
  int i = blockIdx.x * blockDim.x + threadIdx.x;
  if (i >= 9 * C_ * C_) return;
  int ci  = i & (C_ - 1);
  int co  = (i >> 8) & (C_ - 1);
  int khw = i >> 16;
  float v = w[(co * C_ + ci) * 9 + khw];
  wb[i] = (v > 0.f) ? 1 : ((v < 0.f) ? -1 : 0);
}

__global__ __launch_bounds__(256) void k_binconv_wmma(
    const char* __restrict__ xb, const char* __restrict__ wb,
    const char* __restrict__ zpage,
    int* __restrict__ y, float* __restrict__ sum, float* __restrict__ sq) {
  __shared__ __align__(16) char lds_a[2][128 * LDSP];
  __shared__ __align__(16) char lds_b[2][128 * LDSP];

  const int t    = threadIdx.x;
  const int lane = t & 31;
  const int wave = t >> 5;
  const int wm   = wave & 3;
  const int wn   = wave >> 2;
  const int hi   = (lane >> 4) & 1;
  const int ln   = lane & 15;

  const int m0  = blockIdx.x * 128;
  const int co0 = blockIdx.y * 128;

  const int sr = t >> 1;
  const int sq_ = (t & 1) * 32;
  const int ap   = m0 + sr;
  const int a_ow = ap % W_;
  const int a_oh = (ap / W_) % H_;
  const int a_n  = ap / HW_;
  const int base0 = ((a_n * H_ + a_oh) * W_ + a_ow) * C_ + sq_;
  const int bb0   = (co0 + sr) * C_ + sq_;

  int ldh = -1, ldw = -1, lci0 = 0, lkofs = 0;
  int ldoff = -(W_ + 1) * C_;

  auto advance = [&]() {
    lci0 += 64; lkofs += 64;
    if (lci0 == 256) {
      lci0 = 0;
      lkofs += C_ * C_ - 256;
      if (ldw == 1) { ldw = -1; ++ldh; ldoff += (W_ - 2) * C_; }
      else          { ++ldw;          ldoff += C_; }
    }
  };

  auto src_a = [&]() -> const char* {
    const int ih = a_oh + ldh, iw = a_ow + ldw;
    const bool inb = ((unsigned)ih < (unsigned)H_) && ((unsigned)iw < (unsigned)W_);
    return inb ? (xb + base0 + ldoff + lci0) : zpage;
  };

  v8i acc[2][4];
  #pragma unroll
  for (int mt = 0; mt < 2; ++mt)
    #pragma unroll
    for (int nt = 0; nt < 4; ++nt)
      acc[mt][nt] = (v8i){0,0,0,0,0,0,0,0};

#if HAVE_ASYNC_LDS
  auto stage = [&](int buf) {
    g4p ga = (g4p)src_a();
    g4p gb = (g4p)(wb + lkofs + bb0);
    l4p la = (l4p)&lds_a[buf][sr * LDSP + sq_];
    l4p lb = (l4p)&lds_b[buf][sr * LDSP + sq_];
    __builtin_amdgcn_global_load_async_to_lds_b128(ga, la, 0, 0);
    __builtin_amdgcn_global_load_async_to_lds_b128(ga, la, 16, 0);
    __builtin_amdgcn_global_load_async_to_lds_b128(gb, lb, 0, 0);
    __builtin_amdgcn_global_load_async_to_lds_b128(gb, lb, 16, 0);
  };
  auto wait_async = [&]() {
#if __has_builtin(__builtin_amdgcn_s_wait_asynccnt)
    __builtin_amdgcn_s_wait_asynccnt(0);
#else
    asm volatile("s_wait_asynccnt 0x0" ::: "memory");
#endif
  };
#else
  int4 ra0, ra1, rb0, rb1;
  auto stage_load = [&]() {
    const char* ga = src_a();
    const char* gb = wb + lkofs + bb0;
    ra0 = *(const int4*)(ga);
    ra1 = *(const int4*)(ga + 16);
    rb0 = *(const int4*)(gb);
    rb1 = *(const int4*)(gb + 16);
  };
  auto store_stage = [&](int buf) {
    *(int4*)(&lds_a[buf][sr * LDSP + sq_])      = ra0;
    *(int4*)(&lds_a[buf][sr * LDSP + sq_ + 16]) = ra1;
    *(int4*)(&lds_b[buf][sr * LDSP + sq_])      = rb0;
    *(int4*)(&lds_b[buf][sr * LDSP + sq_ + 16]) = rb1;
  };
#endif

  auto compute = [&](int buf) {
    const char* La = lds_a[buf];
    const char* Lb = lds_b[buf];
    v8i afr[2], bfr[4];
    #pragma unroll
    for (int mt = 0; mt < 2; ++mt) {
      const char* pa = La + (wm * 32 + mt * 16 + ln) * LDSP + hi * 8;
      int2 q0 = *(const int2*)(pa +  0);
      int2 q1 = *(const int2*)(pa + 16);
      int2 q2 = *(const int2*)(pa + 32);
      int2 q3 = *(const int2*)(pa + 48);
      afr[mt][0] = q0.x; afr[mt][1] = q0.y; afr[mt][2] = q1.x; afr[mt][3] = q1.y;
      afr[mt][4] = q2.x; afr[mt][5] = q2.y; afr[mt][6] = q3.x; afr[mt][7] = q3.y;
    }
    #pragma unroll
    for (int nt = 0; nt < 4; ++nt) {
      const char* pb = Lb + (wn * 64 + nt * 16 + ln) * LDSP + hi * 8;
      int2 q0 = *(const int2*)(pb +  0);
      int2 q1 = *(const int2*)(pb + 16);
      int2 q2 = *(const int2*)(pb + 32);
      int2 q3 = *(const int2*)(pb + 48);
      bfr[nt][0] = q0.x; bfr[nt][1] = q0.y; bfr[nt][2] = q1.x; bfr[nt][3] = q1.y;
      bfr[nt][4] = q2.x; bfr[nt][5] = q2.y; bfr[nt][6] = q3.x; bfr[nt][7] = q3.y;
    }
    #pragma unroll
    for (int mt = 0; mt < 2; ++mt)
      #pragma unroll
      for (int nt = 0; nt < 4; ++nt)
        acc[mt][nt] = __builtin_amdgcn_wmma_i32_16x16x64_iu8(
            true, afr[mt], true, bfr[nt], acc[mt][nt], false, false);
  };

#if HAVE_ASYNC_LDS
  stage(0);  advance();  wait_async();
  for (int step = 0; step < 36; ++step) {
    __syncthreads();
    const bool more = (step < 35);
    if (more) { stage((step + 1) & 1); advance(); }
    compute(step & 1);
    if (more) wait_async();
  }
#else
  stage_load();  advance();  store_stage(0);
  for (int step = 0; step < 36; ++step) {
    __syncthreads();
    const bool more = (step < 35);
    if (more) { stage_load(); advance(); }
    compute(step & 1);
    if (more) store_stage((step + 1) & 1);
  }
#endif

  #pragma unroll
  for (int mt = 0; mt < 2; ++mt) {
    #pragma unroll
    for (int nt = 0; nt < 4; ++nt) {
      v8i a = acc[mt][nt];
      const int prow = m0 + wm * 32 + mt * 16 + 8 * hi;
      const int ch   = co0 + wn * 64 + nt * 16 + ln;
      float s = 0.f, ss = 0.f;
      #pragma unroll
      for (int r = 0; r < 8; ++r) {
        int v = a[r];
        y[(prow + r) * C_ + ch] = v;
        float f = SC_ * (float)v;
        s  += f;
        ss += f * f;
      }
      s  += __shfl_xor(s, 16, 32);
      ss += __shfl_xor(ss, 16, 32);
      if (hi == 0) {
        atomicAdd(&sum[ch], s);
        atomicAdd(&sq[ch],  ss);
      }
    }
  }
}

__global__ void k_bn_htanh_bin(const int* __restrict__ y,
                               const float* __restrict__ sum,
                               const float* __restrict__ sq,
                               const float* __restrict__ g,
                               const float* __restrict__ b,
                               char* __restrict__ out) {
  int i = (blockIdx.x * blockDim.x + threadIdx.x) * 4;
  if (i >= NHW_ * C_) return;
  int c = i & (C_ - 1);
  int4 v = *(const int4*)(y + i);
  int vv[4] = {v.x, v.y, v.z, v.w};
  unsigned packed = 0;
  #pragma unroll
  for (int j = 0; j < 4; ++j) {
    int cc = c + j;
    float mean = sum[cc] * (1.0f / CNTF);
    float var  = sq[cc] * (1.0f / CNTF) - mean * mean;
    float inv  = rsqrtf(var + EPS_);
    float yv   = (SC_ * (float)vv[j] - mean) * inv * g[cc] + b[cc];
    int sgn = (yv > 0.f) ? 1 : ((yv < 0.f) ? -1 : 0);
    packed |= ((unsigned)(sgn & 0xff)) << (8 * j);
  }
  *(unsigned*)(out + i) = packed;
}

__global__ void k_bn_res_out(const int* __restrict__ y,
                             const float* __restrict__ sum,
                             const float* __restrict__ sq,
                             const float* __restrict__ g,
                             const float* __restrict__ b,
                             const float* __restrict__ x,
                             float* __restrict__ out) {
  int i = blockIdx.x * blockDim.x + threadIdx.x;
  if (i >= N_ * C_ * HW_) return;
  int w = i % W_;
  int h = (i / W_) % H_;
  int c = (i / HW_) % C_;
  int n = i / (C_ * HW_);
  float mean = sum[c] * (1.0f / CNTF);
  float var  = sq[c] * (1.0f / CNTF) - mean * mean;
  float inv  = rsqrtf(var + EPS_);
  int acc = y[((n * H_ + h) * W_ + w) * C_ + c];
  float v = (SC_ * (float)acc - mean) * inv * g[c] + b[c] + x[i];
  out[i] = fminf(1.f, fmaxf(-1.f, v));
}

extern "C" void kernel_launch(void* const* d_in, const int* in_sizes, int n_in,
                              void* d_out, int out_size, void* d_ws, size_t ws_size,
                              hipStream_t stream) {
  (void)in_sizes; (void)n_in; (void)out_size; (void)ws_size;
  const float* x  = (const float*)d_in[0];
  const float* w1 = (const float*)d_in[1];
  const float* g1 = (const float*)d_in[2];
  const float* b1 = (const float*)d_in[3];
  const float* w2 = (const float*)d_in[4];
  const float* g2 = (const float*)d_in[5];
  const float* b2 = (const float*)d_in[6];
  float* out = (float*)d_out;

  char*  ws    = (char*)d_ws;
  char*  xbin  = ws + OFF_XBIN;
  char*  a2bin = ws + OFF_A2BIN;
  char*  wb1   = ws + OFF_WB1;
  char*  wb2   = ws + OFF_WB2;
  float* stats = (float*)(ws + OFF_STATS);
  char*  zpage = ws + OFF_ZERO;
  float* sum1 = stats;        float* sq1 = stats + 256;
  float* sum2 = stats + 512;  float* sq2 = stats + 768;
  int*   y32  = (int*)(ws + OFF_Y32);

  k_init<<<5, 256, 0, stream>>>(stats, (int*)zpage);
  k_binarize_x<<<(NHW_ * C_) / 256, 256, 0, stream>>>(x, xbin);
  k_binarize_w<<<(9 * C_ * C_) / 256, 256, 0, stream>>>(w1, wb1);
  k_binarize_w<<<(9 * C_ * C_) / 256, 256, 0, stream>>>(w2, wb2);

  dim3 grid(NHW_ / 128, C_ / 128);
  k_binconv_wmma<<<grid, 256, 0, stream>>>(xbin, wb1, zpage, y32, sum1, sq1);
  k_bn_htanh_bin<<<(NHW_ * C_ / 4) / 256, 256, 0, stream>>>(y32, sum1, sq1, g1, b1, a2bin);
  k_binconv_wmma<<<grid, 256, 0, stream>>>(a2bin, wb2, zpage, y32, sum2, sq2);
  k_bn_res_out<<<(N_ * C_ * HW_) / 256, 256, 0, stream>>>(y32, sum2, sq2, g2, b2, x, out);
}
